// RTDETRDecoder_33285996544312
// MI455X (gfx1250) — matrix operand order repaired and checked by reading, not compiled, not run on hardware
//
#include <hip/hip_runtime.h>
#include <hip/hip_bf16.h>

// ---------------------------------------------------------------------------
// RT-DETR decoder for MI455X (gfx1250). All large GEMMs run on
// v_wmma_f32_16x16x32_f16 (fp32 in/out, f16 multiply, f32 accumulate).
// wave32 everywhere.
// ---------------------------------------------------------------------------

typedef __attribute__((ext_vector_type(16))) _Float16 v16h;
typedef __attribute__((ext_vector_type(4)))  _Float16 v4h;
typedef __attribute__((ext_vector_type(8)))  float    v8f;

#define LN_EPS 1e-5f

static __device__ __forceinline__ float sigmoidf_(float x) {
    return 1.0f / (1.0f + __expf(-x));
}

// ---------------------------------------------------------------------------
// Generic WMMA GEMM:  C[m,n] = act( sum_k A[m,k]*B[k,n] + bias[n] )
//   A element at A[m*Asm + k*Ask]       (fp32)
//   B element at B[k*Bsk + n*Bsn]       (fp32)
//   C element at C[m*Csm + n]           (fp32)
// Batched via grid.z:  off = (z/zdiv)*Outer + (z%zdiv)*Inner  per operand.
// Optional per-row scale on A (valid-mask fusion): A *= rowScale[m % rsMod].
// Block = 128 threads = 4 waves. Block tile = 32(M) x 64(N).
// The 32x32 A tile is staged once in LDS (f16); each wave owns one 16-column
// strip and TWO 16x16 accumulators, so each B fragment feeds two WMMAs.
// Fast path (row-major A, aligned, in-bounds): float4 loads -> ds_store_b64.
// ---------------------------------------------------------------------------
__global__ __launch_bounds__(128)
void k_gemm(const float* __restrict__ A, const float* __restrict__ B,
            const float* __restrict__ bias, float* __restrict__ C,
            int M, int N, int K,
            long long Asm, long long Ask,
            long long Bsk, long long Bsn,
            long long Csm,
            int zdiv,
            long long aO, long long aI,
            long long bO, long long bI,
            long long cO, long long cI,
            int relu,
            const float* __restrict__ rowScale, int rsMod)
{
    __shared__ _Float16 As[32][32];

    const int z  = blockIdx.z;
    const long long zo = (long long)(z / zdiv);
    const long long zi = (long long)(z % zdiv);
    A += zo * aO + zi * aI;
    B += zo * bO + zi * bI;
    C += zo * cO + zi * cI;

    const int wave  = threadIdx.x >> 5;
    const int lane  = threadIdx.x & 31;
    const int m0    = blockIdx.y * 32;
    const int n0    = (blockIdx.x * 4 + wave) * 16;
    const int mrow  = lane & 15;       // C/D: n = lane&15 ; A: m = lane&15
    const int lhalf = lane >> 4;       // lane group 0/1
    const int ncol  = n0 + (lane & 15);
    const bool aRowMajor = (Ask == 1);
    const bool hasRS = (rowScale != nullptr);
    const bool nOK   = (ncol < N);
    // block-uniform fast-path eligibility for A staging (16B-aligned rows)
    const bool fastA = aRowMajor && ((Asm & 3) == 0) && !hasRS && (m0 + 32 <= M);

    v8f acc0 = {};
    v8f acc1 = {};

    for (int k0 = 0; k0 < K; k0 += 32) {
        __syncthreads();               // previous-iteration readers done
        if (fastA && (k0 + 32 <= K)) {
            // ---- vectorized stage: 256 float4 loads by 128 threads -----
            #pragma unroll
            for (int t0 = 0; t0 < 256; t0 += 128) {
                const int t   = t0 + threadIdx.x;
                const int mm  = t >> 3;
                const int kk4 = (t & 7) << 2;
                const float4 q = *reinterpret_cast<const float4*>(
                    &A[(long long)(m0 + mm) * Asm + (k0 + kk4)]);
                v4h h;
                h[0] = (_Float16)q.x; h[1] = (_Float16)q.y;
                h[2] = (_Float16)q.z; h[3] = (_Float16)q.w;
                *reinterpret_cast<v4h*>(&As[mm][kk4]) = h;
            }
            if (k0 + 32 < K)
                __builtin_prefetch(&A[(long long)(m0 + (threadIdx.x & 31)) * Asm +
                                      (k0 + 32)], 0, 0);
        } else {
            // ---- guarded scalar stage ----------------------------------
            #pragma unroll
            for (int t0 = 0; t0 < 1024; t0 += 128) {
                const int t = t0 + threadIdx.x;
                int mm, kk;
                if (aRowMajor) { mm = t >> 5;  kk = t & 31; }  // coalesce along k
                else           { mm = t & 31;  kk = t >> 5; }  // coalesce along m
                const int gm = m0 + mm;
                const int gk = k0 + kk;
                float v = 0.0f;
                if (gm < M && gk < K) {
                    v = A[(long long)gm * Asm + (long long)gk * Ask];
                    if (hasRS) v *= rowScale[gm % rsMod];
                }
                As[mm][kk] = (_Float16)v;
            }
        }
        __syncthreads();

        // ---- A fragments for both M-tiles (contiguous k = 16*lhalf + j, matches B) --
        v16h a0, a1;
        #pragma unroll
        for (int j = 0; j < 16; ++j) {
            const int v4 = j >> 1;
            const int kk = (v4 < 4) ? (lhalf * 16 + j) : (lhalf * 16 + j);
            a0[j] = As[mrow][kk];
            a1[j] = As[mrow + 16][kk];
        }

        // ---- B fragment (contiguous k = 16*lhalf + j, matches A) -------------------
        v16h bfrag;
        if (nOK && (k0 + 32 <= K)) {
            const float* Bp = B + (long long)(k0 + lhalf * 16) * Bsk +
                              (long long)ncol * Bsn;
            #pragma unroll
            for (int j = 0; j < 16; ++j)
                bfrag[j] = (_Float16)Bp[(long long)j * Bsk];
        } else {
            #pragma unroll
            for (int j = 0; j < 16; ++j) {
                const int gk = k0 + lhalf * 16 + j;
                float v = 0.0f;
                if (gk < K && nOK)
                    v = B[(long long)gk * Bsk + (long long)ncol * Bsn];
                bfrag[j] = (_Float16)v;
            }
        }
        if (k0 + 32 < K && nOK)
            __builtin_prefetch(&B[(long long)(k0 + 32 + lhalf * 16) * Bsk +
                                  (long long)ncol * Bsn], 0, 0);

        acc0 = __builtin_amdgcn_wmma_f32_16x16x32_f16(
                   false, a0, false, bfrag, (short)0, acc0, false, false);
        acc1 = __builtin_amdgcn_wmma_f32_16x16x32_f16(
                   false, a1, false, bfrag, (short)0, acc1, false, false);
    }

    // ---- epilogue (C/D layout: VGPR r -> m = mbase + r + 8*lhalf) ------
    if (nOK) {
        const float bv = bias ? bias[ncol] : 0.0f;
        #pragma unroll
        for (int r = 0; r < 8; ++r) {
            const int m = m0 + r + lhalf * 8;
            if (m < M) {
                float v = acc0[r] + bv;
                if (relu) v = fmaxf(v, 0.0f);
                C[(long long)m * Csm + ncol] = v;
            }
        }
        #pragma unroll
        for (int r = 0; r < 8; ++r) {
            const int m = m0 + 16 + r + lhalf * 8;
            if (m < M) {
                float v = acc1[r] + bv;
                if (relu) v = fmaxf(v, 0.0f);
                C[(long long)m * Csm + ncol] = v;
            }
        }
    }
}

// ---------------------------------------------------------------------------
// BatchNorm-style epilogue for the input projections (per-level).
// F layout: (b*8400 + start + pos)*256 + c
// ---------------------------------------------------------------------------
__global__ void k_bn(float* __restrict__ F, int start, int HW,
                     const float* __restrict__ g, const float* __restrict__ b,
                     const float* __restrict__ m, const float* __restrict__ v)
{
    long long t = (long long)blockIdx.x * 256 + threadIdx.x;
    long long tot = 8ll * HW * 256;
    if (t >= tot) return;
    int c = (int)(t & 255);
    long long r = t >> 8;
    int bb = (int)(r / HW);
    long long pos = r % HW;
    float* p = F + ((long long)bb * 8400 + start + pos) * 256 + c;
    *p = (*p - m[c]) * rsqrtf(v[c] + 1e-5f) * g[c] + b[c];
}

// ---------------------------------------------------------------------------
// Anchor generation (8400 positions across 3 levels) + validity mask.
// ---------------------------------------------------------------------------
__global__ void k_anchors(float* __restrict__ anch, float* __restrict__ valid)
{
    int p = blockIdx.x * 256 + threadIdx.x;
    if (p >= 8400) return;
    int lvl, W, H, st;
    if (p < 6400)      { lvl = 0; W = 80; H = 80; st = 0;    }
    else if (p < 8000) { lvl = 1; W = 40; H = 40; st = 6400; }
    else               { lvl = 2; W = 20; H = 20; st = 8000; }
    int idx = p - st;
    int gy = idx / W, gx = idx % W;
    float a[4];
    a[0] = (gx + 0.5f) / (float)W;
    a[1] = (gy + 0.5f) / (float)H;
    a[2] = 0.05f * (float)(1 << lvl);
    a[3] = a[2];
    bool ok = true;
    #pragma unroll
    for (int j = 0; j < 4; ++j) ok = ok && (a[j] > 0.01f) && (a[j] < 0.99f);
    #pragma unroll
    for (int j = 0; j < 4; ++j)
        anch[p * 4 + j] = ok ? logf(a[j] / (1.0f - a[j])) : 1e9f;
    valid[p] = ok ? 1.0f : 0.0f;
}

// ---------------------------------------------------------------------------
// LayerNorm over 256 with optional residual: out = LN(x [+ res]) * g + b.
// One wave32 per row.
// ---------------------------------------------------------------------------
__global__ __launch_bounds__(256)
void k_ln(const float* __restrict__ x, const float* __restrict__ res,
          const float* __restrict__ g, const float* __restrict__ b,
          float* __restrict__ out, int rows)
{
    int row = blockIdx.x * 8 + (threadIdx.x >> 5);
    if (row >= rows) return;
    int lane = threadIdx.x & 31;
    long long base = (long long)row * 256;
    float vals[8];
    float s = 0.0f;
    #pragma unroll
    for (int i = 0; i < 8; ++i) {
        float v = x[base + lane + i * 32];
        if (res) v += res[base + lane + i * 32];
        vals[i] = v; s += v;
    }
    #pragma unroll
    for (int o = 16; o > 0; o >>= 1) s += __shfl_xor(s, o, 32);
    float mean = s * (1.0f / 256.0f);
    float vs = 0.0f;
    #pragma unroll
    for (int i = 0; i < 8; ++i) { float d = vals[i] - mean; vs += d * d; }
    #pragma unroll
    for (int o = 16; o > 0; o >>= 1) vs += __shfl_xor(vs, o, 32);
    float inv = rsqrtf(vs * (1.0f / 256.0f) + LN_EPS);
    #pragma unroll
    for (int i = 0; i < 8; ++i) {
        int c = lane + i * 32;
        out[base + c] = (vals[i] - mean) * inv * g[c] + b[c];
    }
}

// ---------------------------------------------------------------------------
// Row max over `len` columns (class-score max).
// ---------------------------------------------------------------------------
__global__ void k_rowmax(const float* __restrict__ x, float* __restrict__ out,
                         int rows, int len)
{
    int r = blockIdx.x * 256 + threadIdx.x;
    if (r >= rows) return;
    const float* p = x + (long long)r * len;
    float m = p[0];
    for (int i = 1; i < len; ++i) m = fmaxf(m, p[i]);
    out[r] = m;
}

// ---------------------------------------------------------------------------
// Deterministic top-300 per batch (descending, ties -> lower index first,
// matching jax.lax.top_k). One block per batch; destroys smax.
// ---------------------------------------------------------------------------
__global__ __launch_bounds__(256)
void k_topk(float* __restrict__ smax, int* __restrict__ topk)
{
    __shared__ float bv[256];
    __shared__ int   bi[256];
    float* s = smax + (long long)blockIdx.x * 8400;
    int tid = threadIdx.x;
    for (int it = 0; it < 300; ++it) {
        float best = -3.4e38f; int besti = 0x7fffffff;
        for (int i = tid; i < 8400; i += 256) {
            float v = s[i];
            if (v > best || (v == best && i < besti)) { best = v; besti = i; }
        }
        bv[tid] = best; bi[tid] = besti;
        __syncthreads();
        for (int st = 128; st > 0; st >>= 1) {
            if (tid < st) {
                if (bv[tid + st] > bv[tid] ||
                    (bv[tid + st] == bv[tid] && bi[tid + st] < bi[tid])) {
                    bv[tid] = bv[tid + st]; bi[tid] = bi[tid + st];
                }
            }
            __syncthreads();
        }
        if (tid == 0) { topk[blockIdx.x * 300 + it] = bi[0]; s[bi[0]] = -3.4e38f; }
        __syncthreads();
    }
}

// ---------------------------------------------------------------------------
// Gathers.
// ---------------------------------------------------------------------------
__global__ void k_gather_feat(const float* __restrict__ E,
                              const int* __restrict__ topk,
                              float* __restrict__ embed)
{
    long long t = (long long)blockIdx.x * 256 + threadIdx.x;
    if (t >= 2400ll * 256) return;
    int c = (int)(t & 255);
    int row = (int)(t >> 8);
    int b = row / 300;
    int src = topk[row];
    embed[t] = E[((long long)b * 8400 + src) * 256 + c];
}

__global__ void k_gather_anchor(const float* __restrict__ anch,
                                const int* __restrict__ topk,
                                float* __restrict__ at)
{
    int t = blockIdx.x * 256 + threadIdx.x;
    if (t >= 9600) return;
    int c = t & 3, row = t >> 2;
    at[t] = anch[topk[row] * 4 + c];
}

// ---------------------------------------------------------------------------
// Elementwise helpers.
// ---------------------------------------------------------------------------
__global__ void k_add(const float* __restrict__ a, const float* __restrict__ b,
                      float* __restrict__ o, long long n)
{
    long long t = (long long)blockIdx.x * 256 + threadIdx.x;
    if (t < n) o[t] = a[t] + b[t];
}

__global__ void k_sigmoid_add(const float* __restrict__ a,
                              const float* __restrict__ b,
                              float* __restrict__ o, long long n)
{
    long long t = (long long)blockIdx.x * 256 + threadIdx.x;
    if (t < n) o[t] = sigmoidf_(a[t] + b[t]);
}

// refer = sigmoid(bbox + inv_sig(refer))   (safe in place: same-index RMW)
__global__ void k_refine(const float* __restrict__ bbox,
                         float* __restrict__ refer, long long n)
{
    long long t = (long long)blockIdx.x * 256 + threadIdx.x;
    if (t >= n) return;
    float r = refer[t];
    r = fminf(fmaxf(r, 0.0f), 1.0f);
    float num = fmaxf(r, 1e-5f);
    float den = fmaxf(1.0f - r, 1e-5f);
    refer[t] = sigmoidf_(bbox[t] + logf(num / den));
}

// ---------------------------------------------------------------------------
// Softmax over `len` (<=320) columns, scaled logits. One wave per row.
// ---------------------------------------------------------------------------
__global__ __launch_bounds__(256)
void k_softmax(float* __restrict__ x, int rows, int len, float scale)
{
    int row = blockIdx.x * 8 + (threadIdx.x >> 5);
    if (row >= rows) return;
    int lane = threadIdx.x & 31;
    float* r = x + (long long)row * len;
    float vals[10];
    int cnt = 0;
    float mx = -3.4e38f;
    for (int i = lane; i < len; i += 32) {
        float v = r[i] * scale;
        vals[cnt++] = v;
        mx = fmaxf(mx, v);
    }
    #pragma unroll
    for (int o = 16; o > 0; o >>= 1) mx = fmaxf(mx, __shfl_xor(mx, o, 32));
    float s = 0.0f;
    for (int c = 0; c < cnt; ++c) { vals[c] = __expf(vals[c] - mx); s += vals[c]; }
    #pragma unroll
    for (int o = 16; o > 0; o >>= 1) s += __shfl_xor(s, o, 32);
    float inv = 1.0f / s;
    cnt = 0;
    for (int i = lane; i < len; i += 32) r[i] = vals[cnt++] * inv;
}

// Deformable-attention weight softmax: groups of 12 contiguous values.
__global__ void k_softmax12(float* __restrict__ x, int groups)
{
    int t = blockIdx.x * 256 + threadIdx.x;
    if (t >= groups) return;
    float* r = x + (long long)t * 12;
    float mx = r[0];
    #pragma unroll
    for (int i = 1; i < 12; ++i) mx = fmaxf(mx, r[i]);
    float s = 0.0f;
    float e[12];
    #pragma unroll
    for (int i = 0; i < 12; ++i) { e[i] = __expf(r[i] - mx); s += e[i]; }
    float inv = 1.0f / s;
    #pragma unroll
    for (int i = 0; i < 12; ++i) r[i] = e[i] * inv;
}

// ---------------------------------------------------------------------------
// Deformable sampling. One wave per (b,q,h); lane = channel d (0..31) so the
// four corner gathers are fully coalesced along the 256-channel axis.
// Vv: (b, pos(8400), 256=h*32+d);  samp: (row, 256=h*32+d)
// ---------------------------------------------------------------------------
__global__ __launch_bounds__(256)
void k_deform(const float* __restrict__ Vv, const float* __restrict__ refer,
              const float* __restrict__ offb, const float* __restrict__ awb,
              float* __restrict__ samp)
{
    int wid = blockIdx.x * 8 + (threadIdx.x >> 5);
    if (wid >= 2400 * 8) return;
    int lane = threadIdx.x & 31;
    int h = wid & 7;
    int row = wid >> 3;
    int b = row / 300;

    const int Ws[3] = {80, 40, 20}, Hs[3] = {80, 40, 20}, Ss[3] = {0, 6400, 8000};
    float cx = refer[row * 4 + 0], cy = refer[row * 4 + 1];
    float rw = refer[row * 4 + 2], rh = refer[row * 4 + 3];

    float acc = 0.0f;
    for (int lvl = 0; lvl < 3; ++lvl) {
        const int W = Ws[lvl], H = Hs[lvl];
        const float* vb = Vv + ((long long)b * 8400 + Ss[lvl]) * 256 + h * 32 + lane;
        for (int p = 0; p < 4; ++p) {
            int oi = row * 192 + ((h * 3 + lvl) * 4 + p) * 2;
            float lx = cx + offb[oi + 0] * 0.25f * rw * 0.5f;   // off/NDP * w * 0.5
            float ly = cy + offb[oi + 1] * 0.25f * rh * 0.5f;
            float x = lx * (float)W - 0.5f;
            float y = ly * (float)H - 0.5f;
            float x0 = floorf(x), y0 = floorf(y);
            float dx = x - x0, dy = y - y0;
            int xi = (int)x0, yi = (int)y0;
            bool okx0 = (x0 >= 0.0f) && (x0 <= (float)(W - 1));
            bool okx1 = (x0 + 1.0f >= 0.0f) && (x0 + 1.0f <= (float)(W - 1));
            bool oky0 = (y0 >= 0.0f) && (y0 <= (float)(H - 1));
            bool oky1 = (y0 + 1.0f >= 0.0f) && (y0 + 1.0f <= (float)(H - 1));
            float s = 0.0f;
            if (okx0 && oky0) s += (1 - dx) * (1 - dy) * vb[(long long)(yi * W + xi) * 256];
            if (okx1 && oky0) s += dx * (1 - dy)       * vb[(long long)(yi * W + xi + 1) * 256];
            if (okx0 && oky1) s += (1 - dx) * dy       * vb[(long long)((yi + 1) * W + xi) * 256];
            if (okx1 && oky1) s += dx * dy             * vb[(long long)((yi + 1) * W + xi + 1) * 256];
            acc += awb[row * 96 + h * 12 + lvl * 4 + p] * s;
        }
    }
    samp[(long long)row * 256 + h * 32 + lane] = acc;
}

// ---------------------------------------------------------------------------
// Final writer: out[b,q,0:4] = refined bbox; out[b,q,4:84] = sigmoid(scores).
// ---------------------------------------------------------------------------
__global__ void k_writeout(const float* __restrict__ refer,
                           const float* __restrict__ sc,
                           float* __restrict__ out)
{
    int t = blockIdx.x * 256 + threadIdx.x;
    if (t >= 2400 * 84) return;
    int j = t % 84, row = t / 84;
    out[t] = (j < 4) ? refer[row * 4 + j] : sigmoidf_(sc[row * 80 + (j - 4)]);
}

// ===========================================================================
// Host orchestration
// ===========================================================================
static inline int cdiv(long long a, long long b) { return (int)((a + b - 1) / b); }

static void gemm(hipStream_t s, const float* A, const float* B, const float* bias,
                 float* C, int M, int N, int K,
                 long long Asm, long long Ask, long long Bsk, long long Bsn,
                 long long Csm,
                 int Z, int zdiv,
                 long long aO, long long aI, long long bO, long long bI,
                 long long cO, long long cI,
                 int relu, const float* rowScale, int rsMod)
{
    dim3 g(cdiv(N, 64), cdiv(M, 32), Z);
    k_gemm<<<g, 128, 0, s>>>(A, B, bias, C, M, N, K, Asm, Ask, Bsk, Bsn, Csm,
                             zdiv, aO, aI, bO, bI, cO, cI, relu, rowScale, rsMod);
}

extern "C" void kernel_launch(void* const* d_in, const int* in_sizes, int n_in,
                              void* d_out, int out_size, void* d_ws, size_t ws_size,
                              hipStream_t stream)
{
    (void)in_sizes; (void)n_in; (void)out_size; (void)ws_size;
    hipStream_t s = stream;
#define PW(i) ((const float*)d_in[(i)])
    // Input flattening (setup_inputs insertion order, recursive):
    //  0..2   : x1,x2,x3
    //  3..17  : proj[i]{w,g,b,m,v}
    //  18..23 : enc_out_w/b, enc_ln_g/b, enc_score_w/b
    //  24..29 : enc_bbox (w,b)x3
    //  30..33 : qpos (w,b)x2
    //  34+26L : layer L: {sa_q,sa_k,sa_v,sa_o,off,aw,val,out,ff1,ff2}(w,b), ln1..3(g,b)
    //  190+2i : score_heads ; 202+6i+2j : bbox_heads
    const int LB0 = 34;

    // ---------------- workspace layout (floats) -------------------------
    float* ws = (float*)d_ws;
    float* F     = ws; ws += 17203200ll;   // feats (8*8400 x 256), persists
    float* E     = ws; ws += 17203200ll;   // enc features -> per-layer value buf
    float* S     = ws; ws += 5760000ll;    // scores_all / attn / ffn-hidden
    float* anch  = ws; ws += 33600;
    float* validb= ws; ws += 8400;
    float* smax  = ws; ws += 67200;
    int*   topk  = (int*)ws; ws += 2400;
    float* embed = ws; ws += 614400;
    float* refer = ws; ws += 9600;
    float* anchT = ws; ws += 9600;
    float* qpos  = ws; ws += 614400;
    float* qbuf  = ws; ws += 614400;
    float* Qb    = ws; ws += 614400;
    float* Kb    = ws; ws += 614400;
    float* Vb    = ws; ws += 614400;
    float* Ob    = ws; ws += 614400;
    float* t1    = ws; ws += 614400;
    float* t2    = ws; ws += 614400;
    float* offb  = ws; ws += 460800;
    float* awb   = ws; ws += 230400;
    float* samp  = ws; ws += 614400;
    float* scb   = ws; ws += 192000;
    float* bboxb = ws; ws += 9600;

    // ---------------- input projections + BN ----------------------------
    const int HWs[3] = {6400, 1600, 400};
    const int Cs[3]  = {512, 1024, 2048};
    const int STs[3] = {0, 6400, 8000};
    for (int l = 0; l < 3; ++l) {
        // f[b,pos,o] = sum_c x[b,c,pos] * w[c,o]   (A: Asm=1, Ask=HW)
        gemm(s, PW(l), PW(3 + 5 * l), nullptr, F + (long long)STs[l] * 256,
             HWs[l], 256, Cs[l],
             1, HWs[l], 256, 1, 256,
             8, 1, (long long)Cs[l] * HWs[l], 0, 0, 0, 8400ll * 256, 0,
             0, nullptr, 1);
        k_bn<<<cdiv(8ll * HWs[l] * 256, 256), 256, 0, s>>>(
            F, STs[l], HWs[l], PW(4 + 5 * l), PW(5 + 5 * l), PW(6 + 5 * l), PW(7 + 5 * l));
    }

    // ---------------- anchors + encoder head ----------------------------
    k_anchors<<<cdiv(8400, 256), 256, 0, s>>>(anch, validb);
    // E = (valid * F) @ enc_out_w + b   (valid mask fused as A row-scale)
    gemm(s, F, PW(18), PW(19), E, 67200, 256, 256,
         256, 1, 256, 1, 256, 1, 1, 0, 0, 0, 0, 0, 0, 0, validb, 8400);
    k_ln<<<cdiv(67200, 8), 256, 0, s>>>(E, nullptr, PW(20), PW(21), E, 67200);
    gemm(s, E, PW(22), PW(23), S, 67200, 80, 256,
         256, 1, 80, 1, 80, 1, 1, 0, 0, 0, 0, 0, 0, 0, nullptr, 1);
    k_rowmax<<<cdiv(67200, 256), 256, 0, s>>>(S, smax, 67200, 80);
    k_topk<<<8, 256, 0, s>>>(smax, topk);
    k_gather_feat<<<cdiv(2400ll * 256, 256), 256, 0, s>>>(E, topk, embed);
    k_gather_anchor<<<cdiv(9600, 256), 256, 0, s>>>(anch, topk, anchT);

    // enc bbox MLP: 256 ->relu 256 ->relu 4 ; refer = sigmoid(. + anchor)
    gemm(s, embed, PW(24), PW(25), t1, 2400, 256, 256, 256, 1, 256, 1, 256,
         1, 1, 0, 0, 0, 0, 0, 0, 1, nullptr, 1);
    gemm(s, t1, PW(26), PW(27), t2, 2400, 256, 256, 256, 1, 256, 1, 256,
         1, 1, 0, 0, 0, 0, 0, 0, 1, nullptr, 1);
    gemm(s, t2, PW(28), PW(29), bboxb, 2400, 4, 256, 256, 1, 4, 1, 4,
         1, 1, 0, 0, 0, 0, 0, 0, 0, nullptr, 1);
    k_sigmoid_add<<<cdiv(9600, 256), 256, 0, s>>>(bboxb, anchT, refer, 9600);

    // ---------------- 6 decoder layers -----------------------------------
    const float ATT_SCALE = 0.17677669529663688f;  // 1/sqrt(32)
    for (int i = 0; i < 6; ++i) {
        const int L = LB0 + 26 * i;
        // qpos = MLP(refer): 4 ->relu 512 -> 256
        gemm(s, refer, PW(30), PW(31), S, 2400, 512, 4, 4, 1, 512, 1, 512,
             1, 1, 0, 0, 0, 0, 0, 0, 1, nullptr, 1);
        gemm(s, S, PW(32), PW(33), qpos, 2400, 256, 512, 512, 1, 256, 1, 256,
             1, 1, 0, 0, 0, 0, 0, 0, 0, nullptr, 1);
        // q = embed + qpos
        k_add<<<cdiv(614400, 256), 256, 0, s>>>(embed, qpos, qbuf, 614400);

        // ---- self attention --------------------------------------------
        gemm(s, qbuf,  PW(L + 0), PW(L + 1), Qb, 2400, 256, 256, 256, 1, 256, 1, 256,
             1, 1, 0, 0, 0, 0, 0, 0, 0, nullptr, 1);
        gemm(s, qbuf,  PW(L + 2), PW(L + 3), Kb, 2400, 256, 256, 256, 1, 256, 1, 256,
             1, 1, 0, 0, 0, 0, 0, 0, 0, nullptr, 1);
        gemm(s, embed, PW(L + 4), PW(L + 5), Vb, 2400, 256, 256, 256, 1, 256, 1, 256,
             1, 1, 0, 0, 0, 0, 0, 0, 0, nullptr, 1);
        // attn[z=b*8+h] (300x300) = Q_bh (300x32) @ K_bh^T (32x300)
        gemm(s, Qb, Kb, nullptr, S, 300, 300, 32,
             256, 1, /*Bsk*/1, /*Bsn*/256, 300,
             64, 8, 76800, 32, 76800, 32, 720000, 90000, 0, nullptr, 1);
        k_softmax<<<cdiv(19200, 8), 256, 0, s>>>(S, 19200, 300, ATT_SCALE);
        // O_bh (300x32) = attn @ V_bh
        gemm(s, S, Vb, nullptr, Ob, 300, 32, 300,
             300, 1, 256, 1, 256,
             64, 8, 720000, 90000, 76800, 32, 76800, 32, 0, nullptr, 1);
        gemm(s, Ob, PW(L + 6), PW(L + 7), t1, 2400, 256, 256, 256, 1, 256, 1, 256,
             1, 1, 0, 0, 0, 0, 0, 0, 0, nullptr, 1);
        k_ln<<<cdiv(2400, 8), 256, 0, s>>>(t1, embed, PW(L + 20), PW(L + 21), embed, 2400);

        // ---- deformable cross attention --------------------------------
        k_add<<<cdiv(614400, 256), 256, 0, s>>>(embed, qpos, qbuf, 614400);
        gemm(s, F, PW(L + 12), PW(L + 13), E, 67200, 256, 256, 256, 1, 256, 1, 256,
             1, 1, 0, 0, 0, 0, 0, 0, 0, nullptr, 1);                       // value proj
        gemm(s, qbuf, PW(L + 8),  PW(L + 9),  offb, 2400, 192, 256, 256, 1, 192, 1, 192,
             1, 1, 0, 0, 0, 0, 0, 0, 0, nullptr, 1);
        gemm(s, qbuf, PW(L + 10), PW(L + 11), awb,  2400, 96, 256, 256, 1, 96, 1, 96,
             1, 1, 0, 0, 0, 0, 0, 0, 0, nullptr, 1);
        k_softmax12<<<cdiv(19200, 256), 256, 0, s>>>(awb, 19200);
        k_deform<<<cdiv(19200, 8), 256, 0, s>>>(E, refer, offb, awb, samp);
        gemm(s, samp, PW(L + 14), PW(L + 15), t1, 2400, 256, 256, 256, 1, 256, 1, 256,
             1, 1, 0, 0, 0, 0, 0, 0, 0, nullptr, 1);
        k_ln<<<cdiv(2400, 8), 256, 0, s>>>(t1, embed, PW(L + 22), PW(L + 23), embed, 2400);

        // ---- FFN --------------------------------------------------------
        gemm(s, embed, PW(L + 16), PW(L + 17), S, 2400, 1024, 256, 256, 1, 1024, 1, 1024,
             1, 1, 0, 0, 0, 0, 0, 0, 1, nullptr, 1);
        gemm(s, S, PW(L + 18), PW(L + 19), t1, 2400, 256, 1024, 1024, 1, 256, 1, 256,
             1, 1, 0, 0, 0, 0, 0, 0, 0, nullptr, 1);
        k_ln<<<cdiv(2400, 8), 256, 0, s>>>(t1, embed, PW(L + 24), PW(L + 25), embed, 2400);

        // ---- bbox head + refine ----------------------------------------
        const int BB = 202 + 6 * i;
        gemm(s, embed, PW(BB + 0), PW(BB + 1), t1, 2400, 256, 256, 256, 1, 256, 1, 256,
             1, 1, 0, 0, 0, 0, 0, 0, 1, nullptr, 1);
        gemm(s, t1, PW(BB + 2), PW(BB + 3), t2, 2400, 256, 256, 256, 1, 256, 1, 256,
             1, 1, 0, 0, 0, 0, 0, 0, 1, nullptr, 1);
        gemm(s, t2, PW(BB + 4), PW(BB + 5), bboxb, 2400, 4, 256, 256, 1, 4, 1, 4,
             1, 1, 0, 0, 0, 0, 0, 0, 0, nullptr, 1);
        k_refine<<<cdiv(9600, 256), 256, 0, s>>>(bboxb, refer, 9600);

        if (i == 5) {
            gemm(s, embed, PW(190 + 2 * i), PW(191 + 2 * i), scb, 2400, 80, 256,
                 256, 1, 80, 1, 80, 1, 1, 0, 0, 0, 0, 0, 0, 0, nullptr, 1);
            k_writeout<<<cdiv(2400 * 84, 256), 256, 0, s>>>(refer, scb, (float*)d_out);
        }
    }
#undef PW
}
